// HMultiheadAttentionOpt_25434796326978
// MI455X (gfx1250) — matrix operand order repaired and checked by reading, not compiled, not run on hardware
//
#include <hip/hip_runtime.h>
#include <cmath>

#define E_DIM 512
#define HD 64
#define NH 8
#define EPSF 1e-15f

typedef __attribute__((ext_vector_type(16))) _Float16 v16h;
typedef __attribute__((ext_vector_type(8)))  _Float16 v8h;
typedef __attribute__((ext_vector_type(8)))  float    v8f;

// Build a 16-bit A-fragment (16x32, wave32 layout): elements 0..7 at p, 8..15 at p+8.
__device__ __forceinline__ v16h load_a16(const _Float16* p) {
  union U { v16h v; v8h h[2]; } u;
  u.h[0] = *(const v8h*)(p);
  u.h[1] = *(const v8h*)(p + 8);
  return u.v;
}

// Sum across each 16-lane half of the wave (result in all lanes of the half).
__device__ __forceinline__ float butterfly16(float s) {
  s += __shfl_xor(s, 1);
  s += __shfl_xor(s, 2);
  s += __shfl_xor(s, 4);
  s += __shfl_xor(s, 8);
  return s;
}

// ---------------- prep kernels ----------------

__global__ void cvt_f32_f16(const float* __restrict__ x, _Float16* __restrict__ y, int n) {
  int i = blockIdx.x * blockDim.x + threadIdx.x;
  int stride = gridDim.x * blockDim.x;
  for (; i < n; i += stride) y[i] = (_Float16)x[i];
}

__global__ void transpose_w(const float* __restrict__ W, _Float16* __restrict__ WT, int n) {
  int t = blockIdx.x * blockDim.x + threadIdx.x;
  if (t >= n) return;
  int i = t / E_DIM, o = t % E_DIM;            // coalesced read of W[i][o]
  WT[(size_t)o * E_DIM + i] = (_Float16)W[t];  // WT[out][in]
}

__global__ void col_norms(const float* __restrict__ W, float* __restrict__ zn) {
  int o = blockIdx.x * blockDim.x + threadIdx.x;
  if (o >= E_DIM) return;
  float s = 0.f;
  for (int i = 0; i < E_DIM; ++i) { float v = W[(size_t)i * E_DIM + o]; s += v * v; }
  zn[o] = fmaxf(sqrtf(s), EPSF);
}

__global__ void row_sq(const float* __restrict__ X, float* __restrict__ x2) {
  int row = blockIdx.x, lane = threadIdx.x;
  const float* p = X + (size_t)row * E_DIM;
  float s = 0.f;
  for (int e = lane; e < E_DIM; e += 32) { float v = p[e]; s += v * v; }
  s = butterfly16(s);
  s += __shfl_xor(s, 16);
  if (lane == 0) x2[row] = s;
}

// ---------------- projection: inner = X @ W (WMMA) + hyperbolic h_linear epilogue -------------
// grid = (S/16, 3), block = 256 (8 waves); wave w owns output cols [w*64, w*64+64).

__global__ __launch_bounds__(256) void proj_hlinear(
    const _Float16* __restrict__ X16, const _Float16* __restrict__ WT16,
    const float* __restrict__ zn3, const float* __restrict__ x23,
    const float* __restrict__ bq, const float* __restrict__ bk, const float* __restrict__ bv,
    float* __restrict__ Pf, _Float16* __restrict__ P16, int S)
{
  __shared__ float partsum[8 * 16];
  int y = blockIdx.y;
  int mtile = blockIdx.x;
  int wave = threadIdx.x >> 5, lane = threadIdx.x & 31, lh = lane & 15;
  int hi = lane >> 4;
  int rowoff = hi ? 8 : 0;
  int ka0 = hi ? 16 : 0;    // A-fragment K base per lane-half
  int kb0 = hi ? 16 : 0;    // B-fragment K base per lane-half
  int mbase = mtile * 16;
  int nbase = wave * 64;

  const _Float16* Xb = X16 + (size_t)y * S * E_DIM;
  const _Float16* Wb = WT16 + (size_t)y * E_DIM * E_DIM;
  const float* bias = (y == 0) ? bq : (y == 1) ? bk : bv;
  const float* zn = zn3 + (size_t)y * E_DIM;
  const float* x2 = x23 + (size_t)y * S;

  v8f acc[4] = {};
  const _Float16* arow = Xb + (size_t)(mbase + lh) * E_DIM;
  for (int kk = 0; kk < E_DIM; kk += 32) {
    v16h a = load_a16(arow + kk + ka0);
#pragma unroll
    for (int t = 0; t < 4; ++t) {
      v16h b = *(const v16h*)(Wb + (size_t)(nbase + t * 16 + lh) * E_DIM + kk + kb0);
      acc[t] = __builtin_amdgcn_wmma_f32_16x16x32_f16(false, a, false, b, (short)0,
                                                      acc[t], false, false);
    }
  }

  float x2r[8], lamr[8];
#pragma unroll
  for (int r = 0; r < 8; ++r) {
    x2r[r] = x2[mbase + r + rowoff];
    lamr[r] = 2.f / (1.f - x2r[r]);
  }

  float wvals[4][8];
  float p2[8];
#pragma unroll
  for (int r = 0; r < 8; ++r) p2[r] = 0.f;

#pragma unroll
  for (int t = 0; t < 4; ++t) {
    int col = nbase + t * 16 + lh;
    float znc = zn[col];
    float rb = bias[col];
    float ch = coshf(2.f * rb), sh = sinhf(2.f * rb);
#pragma unroll
    for (int r = 0; r < 8; ++r) {
      float inner = acc[t][r];
      float lam = lamr[r];
      float targ = (inner * lam / znc) * ch - (lam - 1.f) * sh;
      float dd = 2.f * znc * asinhf(targ);
      float wv = sinhf(dd);
      wvals[t][r] = wv;
      p2[r] += wv * wv;
    }
  }
  // row-wise |w|^2 over all 512 cols: reduce 16 lanes then across the 8 waves via LDS
#pragma unroll
  for (int r = 0; r < 8; ++r) {
    float s = butterfly16(p2[r]);
    if (lh == 0) partsum[wave * 16 + r + rowoff] = s;
  }
  __syncthreads();
  float rs[8];
#pragma unroll
  for (int r = 0; r < 8; ++r) {
    float s = 0.f;
#pragma unroll
    for (int w8 = 0; w8 < 8; ++w8) s += partsum[w8 * 16 + r + rowoff];
    rs[r] = 1.f / (1.f + sqrtf(1.f + s));
  }
#pragma unroll
  for (int t = 0; t < 4; ++t) {
    int col = nbase + t * 16 + lh;
    int head = col >> 6, d = col & 63;
#pragma unroll
    for (int r = 0; r < 8; ++r) {
      int row = mbase + r + rowoff;
      float o = wvals[t][r] * rs[r];
      size_t oi = (((size_t)y * NH + head) * S + row) * HD + d;
      Pf[oi] = o;
      P16[oi] = (_Float16)o;
    }
  }
}

// ---------------- per-(head,token) norms: q2, k2, lam_v; V pre-scaled & transposed f16 -------

__global__ void head_norms(const float* __restrict__ Pf, float* __restrict__ q2,
                           float* __restrict__ k2, float* __restrict__ lamv,
                           _Float16* __restrict__ VT16, int S)
{
  int idx = blockIdx.x;   // h*S + s
  int lane = threadIdx.x;
  size_t HS = (size_t)NH * S;
  const float* qrow = Pf + ((size_t)0 * HS + idx) * HD;
  const float* krow = Pf + ((size_t)1 * HS + idx) * HD;
  const float* vrow = Pf + ((size_t)2 * HS + idx) * HD;

  float a0 = qrow[lane], a1 = qrow[lane + 32];
  float s = a0 * a0 + a1 * a1;
  s = butterfly16(s); s += __shfl_xor(s, 16);
  if (lane == 0) q2[idx] = s;

  float b0 = krow[lane], b1 = krow[lane + 32];
  float sk = b0 * b0 + b1 * b1;
  sk = butterfly16(sk); sk += __shfl_xor(sk, 16);
  if (lane == 0) k2[idx] = sk;

  float c0 = vrow[lane], c1 = vrow[lane + 32];
  float sv = c0 * c0 + c1 * c1;
  sv = butterfly16(sv); sv += __shfl_xor(sv, 16);
  float lam = 2.f / (1.f - sv);
  if (lane == 0) lamv[idx] = lam;

  int h = idx / S, ss = idx % S;
  _Float16* vt = VT16 + (size_t)h * HD * S;
  vt[(size_t)lane * S + ss] = (_Float16)(lam * c0);
  vt[(size_t)(lane + 32) * S + ss] = (_Float16)(lam * c1);
}

// ---------------- fused causal hyperbolic attention + gyromidpoint ----------------
// One wave per (head, 16-query-row tile). Q*K^T via WMMA (K=64), arccosh/exp epilogue,
// weight tile through per-wave LDS (layout change C->A), then w @ (lam*V) via WMMA.

__global__ __launch_bounds__(256) void attn_fused(
    const _Float16* __restrict__ Q16, const _Float16* __restrict__ K16,
    const float* __restrict__ q2a, const float* __restrict__ k2a,
    const float* __restrict__ lamva, const _Float16* __restrict__ VT16,
    const float* __restrict__ stau, const float* __restrict__ sgamma,
    float* __restrict__ out, int S, float betaScale)
{
  __shared__ __align__(32) _Float16 ldsw[8 * 16 * 32];
  int wave = threadIdx.x >> 5, lane = threadIdx.x & 31, lh = lane & 15;
  int hi = lane >> 4;
  int rowoff = hi ? 8 : 0, ka0 = hi ? 16 : 0, kb0 = hi ? 16 : 0;
  int MT = S >> 4;
  int task = blockIdx.x * 8 + wave;
  if (task >= NH * MT) return;
  int h = task / MT, mtile = task % MT;
  int mbase = mtile * 16;
  float tau = expf(stau[0]);
  float gamma = sgamma[0];
  _Float16* wl = ldsw + wave * 512;

  const _Float16* Qh = Q16 + (size_t)h * S * HD;
  const _Float16* Kh = K16 + (size_t)h * S * HD;
  const _Float16* Vth = VT16 + (size_t)h * HD * S;
  const float* q2 = q2a + (size_t)h * S;
  const float* k2 = k2a + (size_t)h * S;
  const float* lamv = lamva + (size_t)h * S;

  // Q fragments are invariant across the key loop: preload (K-chunks 0..31, 32..63)
  v16h aq0 = load_a16(Qh + (size_t)(mbase + lh) * HD + 0 + ka0);
  v16h aq1 = load_a16(Qh + (size_t)(mbase + lh) * HD + 32 + ka0);

  float q2r[8], omq[8];
#pragma unroll
  for (int r = 0; r < 8; ++r) {
    q2r[r] = q2[mbase + r + rowoff];
    omq[r] = 1.f - q2r[r];
  }

  v8f accN[4] = {};
  float dacc[8] = {0.f, 0.f, 0.f, 0.f, 0.f, 0.f, 0.f, 0.f};

  int nk = (mbase + 16 + 31) >> 5;  // 32-key steps to cover the causal range
  for (int j32 = 0; j32 < nk; ++j32) {
    int jb = j32 * 32;
    v8f accQK[2];
#pragma unroll
    for (int s2 = 0; s2 < 2; ++s2) {
      const _Float16* kp = Kh + (size_t)(jb + s2 * 16 + lh) * HD + kb0;
      v16h b0 = *(const v16h*)(kp);        // head-dims 0..31 (B layout)
      v16h b1 = *(const v16h*)(kp + 32);   // head-dims 32..63
      v8f z = {};
      v8f t0 = __builtin_amdgcn_wmma_f32_16x16x32_f16(false, aq0, false, b0, (short)0,
                                                      z, false, false);
      accQK[s2] = __builtin_amdgcn_wmma_f32_16x16x32_f16(false, aq1, false, b1, (short)0,
                                                         t0, false, false);
    }
    // epilogue: Poincare distance -> exp attention weight; causal mask; den accumulation
#pragma unroll
    for (int s2 = 0; s2 < 2; ++s2) {
      int j = jb + s2 * 16 + lh;
      float k2j = k2[j];
      float omk = 1.f - k2j;
      float lj = lamv[j] - 1.f;
#pragma unroll
      for (int r = 0; r < 8; ++r) {
        int ig = mbase + r + rowoff;
        float qk = accQK[s2][r];
        float diff2 = fmaxf(q2r[r] + k2j - 2.f * qk, 0.f);
        float arg = 1.f + 2.f * diff2 / (omq[r] * omk);
        float dist = acoshf(fmaxf(arg, 1.f + 1e-7f));
        float wv = (j <= ig) ? expf(-dist * tau - gamma) : 0.f;
        dacc[r] += wv * lj;
        wl[(r + rowoff) * 32 + s2 * 16 + lh] = (_Float16)wv;
      }
    }
    asm volatile("s_wait_dscnt 0" ::: "memory");
    // reload weight tile in A-fragment layout (16 rows x 32 keys)
    v16h aw = load_a16(wl + lh * 32 + ka0);
#pragma unroll
    for (int t = 0; t < 4; ++t) {
      v16h bv = *(const v16h*)(Vth + (size_t)(t * 16 + lh) * S + jb + kb0);
      accN[t] = __builtin_amdgcn_wmma_f32_16x16x32_f16(false, aw, false, bv, (short)0,
                                                       accN[t], false, false);
    }
    asm volatile("s_wait_dscnt 0" ::: "memory");
  }

  // gyromidpoint epilogue
#pragma unroll
  for (int r = 0; r < 8; ++r) {
    float d = butterfly16(dacc[r]);
    dacc[r] = fmaxf(d, EPSF);
  }
#pragma unroll
  for (int r = 0; r < 8; ++r) {
    float g[4];
    float gn2 = 0.f;
#pragma unroll
    for (int t = 0; t < 4; ++t) { g[t] = accN[t][r] / dacc[r]; gn2 += g[t] * g[t]; }
    gn2 = butterfly16(gn2);
    float gn = fmaxf(sqrtf(gn2), EPSF);
    float th = tanhf(0.5f * atanhf(fminf(gn, 1.f - 1e-7f)));
    float f = th / gn * betaScale;
    int row = mbase + r + rowoff;
    float* op = out + (size_t)row * E_DIM + h * HD;
#pragma unroll
    for (int t = 0; t < 4; ++t) op[t * 16 + lh] = f * g[t];
  }
}

// ---------------- host launch ----------------

extern "C" void kernel_launch(void* const* d_in, const int* in_sizes, int n_in,
                              void* d_out, int out_size, void* d_ws, size_t ws_size,
                              hipStream_t stream)
{
  (void)n_in; (void)out_size; (void)ws_size;
  const float* query = (const float*)d_in[0];
  const float* key   = (const float*)d_in[1];
  const float* value = (const float*)d_in[2];
  const float* Wq = (const float*)d_in[3];
  const float* Wk = (const float*)d_in[4];
  const float* Wv = (const float*)d_in[5];
  const float* bq = (const float*)d_in[6];
  const float* bk = (const float*)d_in[7];
  const float* bv = (const float*)d_in[8];
  const float* stau = (const float*)d_in[9];
  const float* sgamma = (const float*)d_in[10];

  const int E = E_DIM;
  const int S = in_sizes[0] / E;
  const int nSE = S * E;

  char* ws = (char*)d_ws;
  size_t off = 0;
  auto take = [&](size_t bytes) -> char* {
    char* p = ws + off;
    off += (bytes + 255) & ~(size_t)255;
    return p;
  };
  _Float16* X16  = (_Float16*)take((size_t)3 * nSE * sizeof(_Float16));
  _Float16* WT16 = (_Float16*)take((size_t)3 * E * E * sizeof(_Float16));
  float* zn3 = (float*)take((size_t)3 * E * sizeof(float));
  float* x23 = (float*)take((size_t)3 * S * sizeof(float));
  float* Pf  = (float*)take((size_t)3 * nSE * sizeof(float));
  _Float16* P16 = (_Float16*)take((size_t)3 * nSE * sizeof(_Float16));
  float* q2 = (float*)take((size_t)NH * S * sizeof(float));
  float* k2 = (float*)take((size_t)NH * S * sizeof(float));
  float* lamv = (float*)take((size_t)NH * S * sizeof(float));
  _Float16* VT16 = (_Float16*)take((size_t)nSE * sizeof(_Float16));

  const float* Xin[3] = {query, key, value};
  const float* Win[3] = {Wq, Wk, Wv};
  for (int y = 0; y < 3; ++y) {
    cvt_f32_f16<<<256, 256, 0, stream>>>(Xin[y], X16 + (size_t)y * nSE, nSE);
    transpose_w<<<(E * E + 255) / 256, 256, 0, stream>>>(Win[y], WT16 + (size_t)y * E * E, E * E);
    col_norms<<<(E + 255) / 256, 256, 0, stream>>>(Win[y], zn3 + (size_t)y * E);
    row_sq<<<S, 32, 0, stream>>>(Xin[y], x23 + (size_t)y * S);
  }

  dim3 gproj(S / 16, 3);
  proj_hlinear<<<gproj, 256, 0, stream>>>(X16, WT16, zn3, x23, bq, bk, bv, Pf, P16, S);
  head_norms<<<NH * S, 32, 0, stream>>>(Pf, q2, k2, lamv, VT16, S);

  // beta-concatenation scale (host math)
  double be = std::exp(std::lgamma(E / 2.0) + std::lgamma(0.5) - std::lgamma(E / 2.0 + 0.5));
  double bh = std::exp(std::lgamma(HD / 2.0) + std::lgamma(0.5) - std::lgamma(HD / 2.0 + 0.5));
  float betaScale = (float)(be / bh);

  _Float16* Q16 = P16;
  _Float16* K16 = P16 + (size_t)nSE;
  int MT = S / 16;
  int nblk = (NH * MT + 7) / 8;
  attn_fused<<<nblk, 256, 0, stream>>>(Q16, K16, q2, k2, lamv, VT16, stau, sgamma,
                                       (float*)d_out, S, betaScale);
}
